// Attention_60112362275180
// MI455X (gfx1250) — matrix operand order repaired and checked by reading, not compiled, not run on hardware
//
#include <hip/hip_runtime.h>
#include <hip/hip_bf16.h>

// MI455X / gfx1250, wave32. All matmuls via v_wmma_f32_16x16x32_bf16.

typedef unsigned short ushort_t;
typedef __attribute__((ext_vector_type(16))) __bf16 v16bf;
typedef __attribute__((ext_vector_type(8)))  float  v8f;

union Frag16 { v16bf v; uint4 q[2]; ushort_t u[16]; unsigned w[8]; };

__device__ __forceinline__ ushort_t f2bf(float f) {
  unsigned u = __builtin_bit_cast(unsigned, f);
  u += 0x7fffu + ((u >> 16) & 1u);           // round-to-nearest-even
  return (ushort_t)(u >> 16);
}

__device__ __forceinline__ v8f wmma_bf16(const Frag16& a, const Frag16& b, v8f c) {
  return __builtin_amdgcn_wmma_f32_16x16x32_bf16(false, a.v, false, b.v,
                                                 (short)0, c, false, false);
}

// DPP16 lane shuffle (stays within each 16-lane half of the wave)
template <int CTRL>
__device__ __forceinline__ float dpp_f(float x) {
  int r = __builtin_amdgcn_update_dpp(0, __builtin_bit_cast(int, x),
                                      CTRL, 0xf, 0xf, true);
  return __builtin_bit_cast(float, r);
}
// max over the 16 lanes of each half: xor1, xor2, xor7(half-mirror), xor15(mirror)
__device__ __forceinline__ float rowmax16(float x) {
  x = fmaxf(x, dpp_f<0x0B1>(x));   // quad_perm(1,0,3,2)
  x = fmaxf(x, dpp_f<0x04E>(x));   // quad_perm(2,3,0,1)
  x = fmaxf(x, dpp_f<0x141>(x));   // row_half_mirror
  x = fmaxf(x, dpp_f<0x140>(x));   // row_mirror
  return x;
}

// Async global->LDS copy, 16B per lane, tracked on ASYNCcnt (CDNA5 TDM-lite path)
__device__ __forceinline__ void async_copy_b128(unsigned lds_off, const void* gaddr) {
  asm volatile("global_load_async_to_lds_b128 %0, %1, off"
               :: "v"(lds_off), "v"(gaddr) : "memory");
}

// ---------------------------------------------------------------------------
// bf16 GEMM: C(4096,1024) = A(4096,1024) * W(1024,1024)
// 256 threads = 8 waves, block tile 128x128, K step 32, double-buffered LDS.
// MODE 0: bf16 out scattered (b,h,t,d)   (Q, K)
// MODE 1: bf16 out scattered (b,h,d,t)   (V transposed)
// MODE 2: f32 out row-major + bias       (output projection)
// ---------------------------------------------------------------------------
template <bool ABF16>
__device__ __forceinline__ void load_tiles(
    ushort_t (&As)[2][128][32], ushort_t (&Bt)[2][128][32], int buf,
    const void* __restrict__ Asrc, const float* __restrict__ Wsrc,
    int m0, int n0, int kk, int ar, int ac, int bk, int bn)
{
  if (ABF16) {
    const ushort_t* ap = (const ushort_t*)Asrc + (size_t)(m0 + ar) * 1024 + kk + ac;
    *(uint4*)&As[buf][ar][ac]     = *(const uint4*)(ap);
    *(uint4*)&As[buf][ar][ac + 8] = *(const uint4*)(ap + 8);
  } else {
    const float* ap = (const float*)Asrc + (size_t)(m0 + ar) * 1024 + kk + ac;
    union { uint4 q[2]; unsigned u[8]; } pk;
    #pragma unroll
    for (int e = 0; e < 8; ++e)
      pk.u[e] = (unsigned)f2bf(ap[2 * e]) | ((unsigned)f2bf(ap[2 * e + 1]) << 16);
    *(uint4*)&As[buf][ar][ac]     = pk.q[0];
    *(uint4*)&As[buf][ar][ac + 8] = pk.q[1];
  }
  const float* wp = Wsrc + (size_t)(kk + bk) * 1024 + n0 + bn;
  #pragma unroll
  for (int e = 0; e < 16; ++e) Bt[buf][bn + e][bk] = f2bf(wp[e]);
}

template <int MODE, bool ABF16>
__global__ __launch_bounds__(256)
void gemm_bf16_kernel(const void* __restrict__ Asrc, const float* __restrict__ Wsrc,
                      const float* __restrict__ bias, void* __restrict__ dst)
{
  __shared__ ushort_t As[2][128][32];   // bf16 A tile [m][k]
  __shared__ ushort_t Bt[2][128][32];   // bf16 W tile transposed [n][k]

  const int tid  = threadIdx.x;
  const int wave = tid >> 5;            // 0..7
  const int lane = tid & 31;
  const int ln   = lane & 15;
  const int hi   = lane >> 4;

  const int m0 = (blockIdx.x >> 3) * 128;   // 32 m-blocks
  const int n0 = (blockIdx.x & 7) * 128;    //  8 n-blocks

  const v8f Z8 = {0.f,0.f,0.f,0.f,0.f,0.f,0.f,0.f};
  v8f C[8] = {Z8, Z8, Z8, Z8, Z8, Z8, Z8, Z8};

  const int ar = tid >> 1;            // 0..127  A-tile row
  const int ac = (tid & 1) * 16;      // 0/16    A-tile col base
  const int bk = tid >> 3;            // 0..31   W-tile k-row
  const int bn = (tid & 7) * 16;      // 0..112  W-tile n base

  load_tiles<ABF16>(As, Bt, 0, Asrc, Wsrc, m0, n0, 0, ar, ac, bk, bn);

  for (int kk = 0; kk < 1024; kk += 32) {
    const int cur = (kk >> 5) & 1;
    __syncthreads();
    if (kk + 32 < 1024)
      load_tiles<ABF16>(As, Bt, cur ^ 1, Asrc, Wsrc, m0, n0, kk + 32, ar, ac, bk, bn);
    if (kk + 64 < 1024) {               // global_prefetch_b8 two steps ahead
      if (ABF16)
        __builtin_prefetch((const ushort_t*)Asrc + (size_t)(m0 + ar) * 1024 + kk + 64 + ac, 0, 3);
      else
        __builtin_prefetch((const float*)Asrc + (size_t)(m0 + ar) * 1024 + kk + 64 + ac, 0, 3);
      __builtin_prefetch(Wsrc + (size_t)(kk + 64 + bk) * 1024 + n0 + bn, 0, 3);
    }

    // A fragment: lane = row m; K {0..15} (lo half) / {16..31} (hi half)
    Frag16 a;
    const int mrow = wave * 16 + ln;
    a.q[0] = *(const uint4*)&As[cur][mrow][hi ? 16 : 0];
    a.q[1] = *(const uint4*)&As[cur][mrow][(hi ? 16 : 0) + 8];

    #pragma unroll
    for (int f = 0; f < 8; ++f) {
      // B fragment: lane = col n; K {0..15} (lo half) / {16..31} (hi half)
      Frag16 b;
      const int ncol = f * 16 + ln;
      b.q[0] = *(const uint4*)&Bt[cur][ncol][hi ? 16 : 0];
      b.q[1] = *(const uint4*)&Bt[cur][ncol][(hi ? 16 : 0) + 8];
      C[f] = wmma_bf16(a, b, C[f]);
    }
  }

  #pragma unroll
  for (int f = 0; f < 8; ++f) {
    #pragma unroll
    for (int r = 0; r < 8; ++r) {
      const int row = m0 + wave * 16 + r + hi * 8;   // C layout
      const int col = n0 + f * 16 + ln;
      const float val = C[f][r];
      if (MODE == 2) {
        ((float*)dst)[(size_t)row * 1024 + col] = val + bias[col];
      } else {
        const int b = row >> 10, t = row & 1023;
        const int h = col >> 6,  d = col & 63;
        size_t off;
        if (MODE == 0) off = (((size_t)(b * 16 + h)) * 1024 + t) * 64 + d;
        else           off = (((size_t)(b * 16 + h)) * 64 + d) * 1024 + t;
        ((ushort_t*)dst)[off] = f2bf(val);
      }
    }
  }
}

// ---------------------------------------------------------------------------
// Flash attention: 4 waves / block, each wave a 16-query tile, 32-key blocks.
// S = 4 WMMAs; online softmax in C-layout (DPP16 row-max, row-sum via an
// extra WMMA against an all-ones B fragment); P via per-wave LDS transpose;
// P*V = 4 WMMAs against pre-transposed V (b,h,d,t).
// ---------------------------------------------------------------------------
__global__ __launch_bounds__(128)
void attn_kernel(const ushort_t* __restrict__ Qb, const ushort_t* __restrict__ Kb,
                 const ushort_t* __restrict__ Vtb, const float* __restrict__ relb,
                 ushort_t* __restrict__ Ob)
{
  __shared__ ushort_t P[4][16][32];             // per-wave P tile, bf16
  __shared__ __align__(16) float biasS[1088];   // bias row + 64-entry overshoot pad

  const int tid  = threadIdx.x;
  const int wave = tid >> 5;
  const int lane = tid & 31;
  const int ln   = lane & 15;
  const int hi   = lane >> 4;

  const int qb = blockIdx.x & 15;     // 16 q-blocks of 64 rows
  const int bh = blockIdx.x >> 4;     // b*16 + h
  const int h  = bh & 15;
  const int b  = bh >> 4;

  // stage bias row via async global->LDS (ASYNCcnt); pad region zeroed.
  {
    const float* src = relb + h * 1024;
    #pragma unroll
    for (int p = 0; p < 2; ++p) {
      const int e = p * 512 + tid * 4;          // 4 floats = 16B per lane
      async_copy_b128((unsigned)(size_t)&biasS[e], src + e);
    }
    if (tid < 64) biasS[1024 + tid] = 0.0f;
    asm volatile("s_wait_asynccnt 0x0" ::: "memory");
    __syncthreads();
  }

  const int i0 = qb * 64 + wave * 16;
  const float inv64 = 1.0f / 64.0f;

  // Q A-fragments, two k=32 chunks over d=64
  Frag16 qa[2];
  {
    const ushort_t* qrow = Qb + ((size_t)bh * 1024 + i0 + ln) * 64;
    #pragma unroll
    for (int c = 0; c < 2; ++c) {
      qa[c].q[0] = *(const uint4*)(qrow + 32 * c + (hi ? 16 : 0));
      qa[c].q[1] = *(const uint4*)(qrow + 32 * c + (hi ? 16 : 0) + 8);
    }
  }

  // constant all-ones B fragment (row-sum column for l)
  Frag16 ones;
  #pragma unroll
  for (int e = 0; e < 8; ++e) ones.w[e] = 0x3f803f80u;   // bf16 1.0 x2

  const v8f Z8 = {0.f,0.f,0.f,0.f,0.f,0.f,0.f,0.f};
  float mrow[8];
  v8f O[5] = {Z8, Z8, Z8, Z8, Z8};    // O[4] accumulates row sums l
  #pragma unroll
  for (int r = 0; r < 8; ++r) mrow[r] = -1e30f;

  const int nkb = ((i0 + 15) >> 5) + 1;   // causal key-block count

  for (int kb = 0; kb < nkb; ++kb) {
    const int j0 = kb * 32;

    // ---- S = Q K^T for 32 keys ----
    v8f S0 = Z8, S1 = Z8;
    #pragma unroll
    for (int c = 0; c < 2; ++c) {
      Frag16 k0, k1;
      const ushort_t* kr0 = Kb + ((size_t)bh * 1024 + j0 + ln)      * 64 + 32 * c + (hi ? 16 : 0);
      const ushort_t* kr1 = Kb + ((size_t)bh * 1024 + j0 + 16 + ln) * 64 + 32 * c + (hi ? 16 : 0);
      k0.q[0] = *(const uint4*)(kr0); k0.q[1] = *(const uint4*)(kr0 + 8);
      k1.q[0] = *(const uint4*)(kr1); k1.q[1] = *(const uint4*)(kr1 + 8);
      S0 = wmma_bf16(qa[c], k0, S0);
      S1 = wmma_bf16(qa[c], k1, S1);
    }

    // ---- online softmax in C-layout (no clamps: biasS is overshoot-padded) ----
    #pragma unroll
    for (int r = 0; r < 8; ++r) {
      const int i    = i0 + r + hi * 8;
      const int key0 = j0 + ln, key1 = key0 + 16;
      const int idx0 = 1023 - i + key0;           // in [0, 1069] subset of [0,1087]
      const float b0 = biasS[idx0], b1 = biasS[idx0 + 16];
      float s0 = (S0[r] + b0) * inv64; s0 = (key0 <= i) ? s0 : -1e30f;
      float s1 = (S1[r] + b1) * inv64; s1 = (key1 <= i) ? s1 : -1e30f;

      const float t     = rowmax16(fmaxf(s0, s1));
      const float mnew  = fmaxf(mrow[r], t);
      const float alpha = __expf(mrow[r] - mnew);
      mrow[r] = mnew;
      #pragma unroll
      for (int f = 0; f < 5; ++f) O[f][r] *= alpha;

      const int m = r + hi * 8;
      P[wave][m][ln]      = f2bf(__expf(s0 - mnew));
      P[wave][m][16 + ln] = f2bf(__expf(s1 - mnew));
    }

    // per-wave LDS ordering: P stores visible before A-layout reads
    asm volatile("s_wait_dscnt 0x0" ::: "memory");

    Frag16 pa;   // P as A-fragment (contraction over 32 keys)
    pa.q[0] = *(const uint4*)&P[wave][ln][hi ? 16 : 0];
    pa.q[1] = *(const uint4*)&P[wave][ln][(hi ? 16 : 0) + 8];

    #pragma unroll
    for (int f = 0; f < 4; ++f) {
      Frag16 vb;  // V^T row = d-column, contiguous over keys
      const ushort_t* vr = Vtb + ((size_t)bh * 64 + f * 16 + ln) * 1024 + j0 + (hi ? 16 : 0);
      vb.q[0] = *(const uint4*)(vr);
      vb.q[1] = *(const uint4*)(vr + 8);
      O[f] = wmma_bf16(pa, vb, O[f]);
    }
    O[4] = wmma_bf16(pa, ones, O[4]);   // l += row-sum(P)
    asm volatile("" ::: "memory");      // keep next iter's P stores after reads
  }

  // normalize + store as (b, t, h*64+d) bf16
  #pragma unroll
  for (int f = 0; f < 4; ++f) {
    #pragma unroll
    for (int r = 0; r < 8; ++r) {
      const int i   = i0 + r + hi * 8;
      const int col = h * 64 + f * 16 + ln;
      Ob[((size_t)b * 1024 + i) * 1024 + col] = f2bf(O[f][r] / O[4][r]);
    }
  }
}

// ---------------------------------------------------------------------------
extern "C" void kernel_launch(void* const* d_in, const int* in_sizes, int n_in,
                              void* d_out, int out_size, void* d_ws, size_t ws_size,
                              hipStream_t stream)
{
  (void)in_sizes; (void)n_in; (void)out_size; (void)ws_size;
  const float* x  = (const float*)d_in[0];
  const float* Wq = (const float*)d_in[1];
  const float* Wk = (const float*)d_in[2];
  const float* Wv = (const float*)d_in[3];
  const float* Wo = (const float*)d_in[4];
  const float* bo = (const float*)d_in[5];
  const float* rb = (const float*)d_in[6];

  const size_t TENS = (size_t)4 * 16 * 1024 * 64;   // 4 Mi bf16 elements
  ushort_t* Qb  = (ushort_t*)d_ws;                  //  8 MB  (b,h,t,d)
  ushort_t* Kb  = Qb  + TENS;                       //  8 MB  (b,h,t,d)
  ushort_t* Vtb = Kb  + TENS;                       //  8 MB  (b,h,d,t)
  ushort_t* Ob  = Vtb + TENS;                       //  8 MB  (b,t,h*dh)

  const dim3 gg(256), gb(256);
  gemm_bf16_kernel<0, false><<<gg, gb, 0, stream>>>(x,  Wq, nullptr, Qb);
  gemm_bf16_kernel<0, false><<<gg, gb, 0, stream>>>(x,  Wk, nullptr, Kb);
  gemm_bf16_kernel<1, false><<<gg, gb, 0, stream>>>(x,  Wv, nullptr, Vtb);
  attn_kernel<<<dim3(1024), dim3(128), 0, stream>>>(Qb, Kb, Vtb, rb, Ob);
  gemm_bf16_kernel<2, true ><<<gg, gb, 0, stream>>>(Ob, Wo, bo, (float*)d_out);
}
